// WaveNet_46514495815767
// MI455X (gfx1250) — matrix operand order repaired and checked by reading, not compiled, not run on hardware
//
#include <hip/hip_runtime.h>
#include <hip/hip_bf16.h>

typedef __attribute__((ext_vector_type(16))) _Float16 v16h;
typedef __attribute__((ext_vector_type(8)))  _Float16 v8h;
typedef __attribute__((ext_vector_type(8)))  float    v8f;

#define LAYERS   30
#define RES_CH   64
#define GATE_CH  128
#define SKIP_CH  64
#define CIN_CH   80
#define CIN_PAD  96
#define OUT_CH   256
#define TPAD     512         // zero pad in front of h (max dilation)
#define TT       128         // time tile per block (layer kernel)
#define NSUB     (TT / 16)
#define NTILE    9           // packed A-fragment tiles per (layer, wave)

__device__ __forceinline__ v8f wmma16(v16h a, v16h b, v8f c) {
  // D(16x16,f32) = A(16x32,f16) x B(32x16,f16) + C
  return __builtin_amdgcn_wmma_f32_16x16x32_f16(false, a, false, b, (short)0, c,
                                                false, false);
}

__device__ __forceinline__ v8f vzero8() {
  v8f z = {0.f, 0.f, 0.f, 0.f, 0.f, 0.f, 0.f, 0.f};
  return z;
}

// Load 16x32 A fragment from f32 row-major weights (row stride ld floats,
// K element stride es floats). K >= kmax reads as zero (pads K).
// K order used here (B loads match): M = lane%16; lanes 0-15 hold K=0..15 (v[0..15]);
// lanes 16-31 hold K=16..31.
__device__ __forceinline__ v16h load_a_f32(const float* __restrict__ W, int ld,
                                           int es, int kmax) {
  const int lane = threadIdx.x & 31;
  const int m = lane & 15;
  const int kb = (lane >> 4) * 16;
  v16h a;
#pragma unroll
  for (int e = 0; e < 16; ++e) {
    int k = kb + e;
    float v = (k < kmax) ? W[(size_t)m * ld + (size_t)k * es] : 0.0f;
    a[e] = (_Float16)v;
  }
  return a;
}

// ------------------------------------------------------------- weight packing
// Pack every A fragment used by the layer kernels into fragment-order f16:
// wpack[((layer*8 + wave)*NTILE + tile)*512 + lane*16 + e].
// Tiles: 0-1 tap0 K0/K1, 2-3 tap1 K0/K1, 4-6 cond K0..K2 (K padded 80->96),
// 7-8 stage-2 (skip_w for waves 0-3, out_w for waves 4-7) K0/K1.
__global__ void k_pack(const float* __restrict__ conv_w,
                       const float* __restrict__ cond_w,
                       const float* __restrict__ skip_w,
                       const float* __restrict__ out_w,
                       _Float16* __restrict__ wpack) {
  int i = blockIdx.x * blockDim.x + threadIdx.x;
  const int total = LAYERS * 8 * NTILE * 32;
  if (i >= total) return;
  const int lane = i & 31;
  const int frag = i >> 5;
  const int tile = frag % NTILE;
  const int wv = (frag / NTILE) & 7;
  const int layer = frag / (NTILE * 8);
  const float* cw = conv_w + (size_t)layer * GATE_CH * RES_CH * 2;
  const float* dw = cond_w + (size_t)layer * GATE_CH * CIN_CH;
  const float* sw = skip_w + (size_t)layer * SKIP_CH * 64;
  const float* ow = out_w + (size_t)layer * RES_CH * 64;
  const int go = wv * 16;
  const float* W;
  int ld, es, kmax;
  switch (tile) {
    case 0: W = cw + (size_t)go * 128 + 0;  ld = 128; es = 2; kmax = 32; break;
    case 1: W = cw + (size_t)go * 128 + 64; ld = 128; es = 2; kmax = 32; break;
    case 2: W = cw + (size_t)go * 128 + 1;  ld = 128; es = 2; kmax = 32; break;
    case 3: W = cw + (size_t)go * 128 + 65; ld = 128; es = 2; kmax = 32; break;
    case 4: W = dw + (size_t)go * CIN_CH;        ld = CIN_CH; es = 1; kmax = 32; break;
    case 5: W = dw + (size_t)go * CIN_CH + 32;   ld = CIN_CH; es = 1; kmax = 32; break;
    case 6: W = dw + (size_t)go * CIN_CH + 64;   ld = CIN_CH; es = 1; kmax = 16; break;
    default: {
      const float* w2 = (wv < 4) ? sw : ow;
      const int m2 = (wv & 3) * 16;
      W = w2 + (size_t)m2 * 64 + (tile == 8 ? 32 : 0);
      ld = 64; es = 1; kmax = 32;
    }
  }
  const int m = lane & 15;
  const int kb = (lane >> 4) * 16;
  _Float16* dst = wpack + (size_t)frag * 512 + lane * 16;
#pragma unroll
  for (int e = 0; e < 16; ++e) {
    int k = kb + e;
    dst[e] = (_Float16)((k < kmax) ? W[(size_t)m * ld + (size_t)k * es] : 0.0f);
  }
}

// ---------------------------------------------------------------- pre kernels
// Activations are TIME-MAJOR: h [B][TPAD+T][64] f16 (front TPAD rows zero),
// c [B][T][96] f16, skips [B][T][64] f32, y1 [B][T][64] f16.

__global__ void k_zero_hpad(_Float16* __restrict__ h, int B, int T) {
  const size_t per_b = (size_t)TPAD * RES_CH;
  const size_t total = (size_t)B * per_b;
  const size_t row = (size_t)(TPAD + T) * RES_CH;
  for (size_t i = (size_t)blockIdx.x * blockDim.x + threadIdx.x; i < total;
       i += (size_t)gridDim.x * blockDim.x) {
    size_t b = i / per_b, off = i % per_b;
    h[b * row + off] = (_Float16)0.0f;
  }
}

__global__ void k_embed(const int* __restrict__ x, const float* __restrict__ fw,
                        const float* __restrict__ fb, _Float16* __restrict__ h,
                        int B, int T) {
  const size_t total = (size_t)B * T;
  for (size_t i = (size_t)blockIdx.x * blockDim.x + threadIdx.x; i < total;
       i += (size_t)gridDim.x * blockDim.x) {
    size_t b = i / T, t = i % T;
    int xi = x[i];
    _Float16* dst = h + ((size_t)b * (TPAD + T) + TPAD + t) * RES_CH;
#pragma unroll 8
    for (int ch = 0; ch < RES_CH; ++ch)
      dst[ch] = (_Float16)(fw[ch * OUT_CH + xi] + fb[ch]);
  }
}

__global__ void k_zero_f32(float* __restrict__ p, size_t n) {
  for (size_t i = (size_t)blockIdx.x * blockDim.x + threadIdx.x; i < n;
       i += (size_t)gridDim.x * blockDim.x)
    p[i] = 0.0f;
}

__global__ void k_zero_f16(_Float16* __restrict__ p, size_t n) {
  for (size_t i = (size_t)blockIdx.x * blockDim.x + threadIdx.x; i < n;
       i += (size_t)gridDim.x * blockDim.x)
    p[i] = (_Float16)0.0f;
}

__global__ void k_cond_proj(const float* __restrict__ cond,
                            const float* __restrict__ cw,
                            float* __restrict__ out, int B, int Tf) {
  const size_t total = (size_t)B * CIN_CH * Tf;
  for (size_t i = (size_t)blockIdx.x * blockDim.x + threadIdx.x; i < total;
       i += (size_t)gridDim.x * blockDim.x) {
    int f = (int)(i % Tf);
    int o = (int)((i / Tf) % CIN_CH);
    int b = (int)(i / ((size_t)Tf * CIN_CH));
    float acc = 0.0f;
    for (int k = 0; k < CIN_CH; ++k)
      acc += cw[o * CIN_CH + k] * cond[((size_t)b * CIN_CH + k) * Tf + f];
    out[i] = acc;  // channel-major intermediate
  }
}

// nearest-upsample by s then box filter of width 2s+1, zero pad s (ch-major)
__global__ void k_upsample_f32(const float* __restrict__ in, float* __restrict__ out,
                               int B, int C, int Tin, int s) {
  const int Tout = Tin * s;
  const float inv = 1.0f / (float)(2 * s + 1);
  const size_t total = (size_t)B * C * Tout;
  for (size_t i = (size_t)blockIdx.x * blockDim.x + threadIdx.x; i < total;
       i += (size_t)gridDim.x * blockDim.x) {
    int t = (int)(i % Tout);
    int c = (int)((i / Tout) % C);
    int b = (int)(i / ((size_t)Tout * C));
    const float* src = in + ((size_t)b * C + c) * Tin;
    float acc = 0.0f;
    for (int j = -s; j <= s; ++j) {
      int u = t + j;
      if (u >= 0 && u < Tout) acc += src[u / s];
    }
    out[i] = acc * inv;
  }
}

// final upsample stage: channel-major f32 in -> TIME-MAJOR padded f16 out
__global__ void k_upsample_f16pad(const float* __restrict__ in,
                                  _Float16* __restrict__ out, int B, int C,
                                  int Tin, int s) {
  const int Tout = Tin * s;
  const float inv = 1.0f / (float)(2 * s + 1);
  const size_t total = (size_t)B * C * Tout;
  for (size_t i = (size_t)blockIdx.x * blockDim.x + threadIdx.x; i < total;
       i += (size_t)gridDim.x * blockDim.x) {
    int t = (int)(i % Tout);
    int c = (int)((i / Tout) % C);
    int b = (int)(i / ((size_t)Tout * C));
    const float* src = in + ((size_t)b * C + c) * Tin;
    float acc = 0.0f;
    for (int j = -s; j <= s; ++j) {
      int u = t + j;
      if (u >= 0 && u < Tout) acc += src[u / s];
    }
    out[((size_t)b * Tout + t) * CIN_PAD + c] = (_Float16)(acc * inv);
  }
}

// ---------------------------------------------------------------- layer kernel

__global__ __launch_bounds__(256) void k_layer(
    const _Float16* __restrict__ hin, _Float16* __restrict__ hout,
    const _Float16* __restrict__ cf16, float* __restrict__ skips,
    const _Float16* __restrict__ wpack, const float* __restrict__ conv_b,
    const float* __restrict__ skip_b, const float* __restrict__ out_b,
    int layer, int T) {
  __shared__ float sh_g[GATE_CH * 16];              // gate exchange (f32)
  __shared__ alignas(32) _Float16 sh_z[16 * 64];    // z, time-major [n][m]

  const int tid = threadIdx.x;
  const int lane = tid & 31;
  const int w = tid >> 5;                           // wave id (8 waves)
  const int b = blockIdx.y;
  const int t0 = blockIdx.x * TT;
  const int d = 1 << (layer % 10);
  const int go = w * 16;
  const int n = lane & 15;
  const int mb = (lane >> 4) * 8;
  const int kb = (lane >> 4) * 16;

  // packed A fragments: 9 contiguous 32B loads per lane
  const _Float16* wp =
      wpack + ((size_t)(layer * 8 + w) * NTILE) * 512 + lane * 16;
  const v16h a_t0k0 = *(const v16h*)(wp + 0 * 512);
  const v16h a_t0k1 = *(const v16h*)(wp + 1 * 512);
  const v16h a_t1k0 = *(const v16h*)(wp + 2 * 512);
  const v16h a_t1k1 = *(const v16h*)(wp + 3 * 512);
  const v16h a_cd0  = *(const v16h*)(wp + 4 * 512);
  const v16h a_cd1  = *(const v16h*)(wp + 5 * 512);
  const v16h a_cd2  = *(const v16h*)(wp + 6 * 512);
  const v16h a_s20  = *(const v16h*)(wp + 7 * 512);
  const v16h a_s21  = *(const v16h*)(wp + 8 * 512);

  // biases hoisted to registers
  float cbv[8], b2v[8];
  const float* cb = conv_b + layer * GATE_CH;
#pragma unroll
  for (int r = 0; r < 8; ++r) cbv[r] = cb[go + mb + r];
  const float* bb2 =
      (w < 4) ? (skip_b + layer * SKIP_CH) : (out_b + layer * RES_CH);
#pragma unroll
  for (int r = 0; r < 8; ++r) b2v[r] = bb2[(w & 3) * 16 + mb + r];

  // per-lane streaming pointers (front pad makes h[t-d] reads branch-free)
  const size_t TSTR = (size_t)(TPAD + T);
  const _Float16* hb = hin + ((size_t)b * TSTR + TPAD) * RES_CH;
  const _Float16* pcur = hb + (size_t)(t0 + n) * RES_CH + kb;
  const _Float16* pprev = pcur - (size_t)d * RES_CH;
  const _Float16* pcnd =
      cf16 + ((size_t)b * T + t0 + n) * CIN_PAD + kb;
  const _Float16* pz = (const _Float16*)sh_z + n * 64 + kb;

  float* sp = skips + ((size_t)b * T + t0 + n) * SKIP_CH + (w & 3) * 16 + mb;
  size_t hoff =
      ((size_t)b * TSTR + TPAD + t0 + n) * RES_CH + (w & 3) * 16 + mb;
  float* shg = sh_g + (go + mb) * 16 + n;

  for (int ts = 0; ts < NSUB; ++ts) {
    // issue all 7 B-fragment loads; the sched_barrier keeps every load ahead
    // of the WMMA chain so all 7 fragments are simultaneously live (distinct
    // registers, one load clause, progressive s_wait_loadcnt)
    const v16h b_p0 = *(const v16h*)(pprev);
    const v16h b_p1 = *(const v16h*)(pprev + 32);
    const v16h b_c0 = *(const v16h*)(pcur);
    const v16h b_c1 = *(const v16h*)(pcur + 32);
    const v16h b_d0 = *(const v16h*)(pcnd);
    const v16h b_d1 = *(const v16h*)(pcnd + 32);
    const v16h b_d2 = *(const v16h*)(pcnd + 64);
    __builtin_amdgcn_sched_barrier(0);

    // gate rows [go, go+16): g = W0*h[t-d] + W1*h[t] + Wc*c
    v8f g = vzero8();
    g = wmma16(a_t0k0, b_p0, g);
    g = wmma16(a_t0k1, b_p1, g);
    g = wmma16(a_t1k0, b_c0, g);
    g = wmma16(a_t1k1, b_c1, g);
    g = wmma16(a_cd0,  b_d0, g);
    g = wmma16(a_cd1,  b_d1, g);
    g = wmma16(a_cd2,  b_d2, g);
#pragma unroll
    for (int r = 0; r < 8; ++r) shg[r * 16] = g[r] + cbv[r];
    __syncthreads();

    // z = tanh(a) * sigmoid(b), via fast exp (v_exp_f32)
#pragma unroll
    for (int ii = 0; ii < 4; ++ii) {
      int i = tid + ii * 256;
      int m = i & 63, nn = i >> 6;
      float av = sh_g[m * 16 + nn];
      float bv = sh_g[(64 + m) * 16 + nn];
      float th = 1.0f - 2.0f / (__expf(2.0f * av) + 1.0f);
      float sg = 1.0f / (1.0f + __expf(-bv));
      sh_z[nn * 64 + m] = (_Float16)(th * sg);
    }
    __syncthreads();

    // stage 2: skip_w @ z (waves 0-3) / out_w @ z (waves 4-7)
    const v16h b_z0 = *(const v16h*)(pz);
    const v16h b_z1 = *(const v16h*)(pz + 32);
    v8f o2 = vzero8();
    o2 = wmma16(a_s20, b_z0, o2);
    o2 = wmma16(a_s21, b_z1, o2);

    if (w < 4) {
      v8f old = *(const v8f*)sp;
#pragma unroll
      for (int r = 0; r < 8; ++r) old[r] += o2[r] + b2v[r];
      *(v8f*)sp = old;
    } else {
      const v8h hold = *(const v8h*)(hin + hoff);
      v8h hv;
#pragma unroll
      for (int r = 0; r < 8; ++r)
        hv[r] = (_Float16)((float)hold[r] + o2[r] + b2v[r]);
      *(v8h*)(hout + hoff) = hv;
    }
    // NOTE: no barrier here -- the next iteration's sh_g/sh_z writes are
    // already separated from this iteration's sh_z reads by barrier 1/2.

    pcur += 16 * RES_CH;
    pprev += 16 * RES_CH;
    pcnd += 16 * CIN_PAD;
    sp += 16 * SKIP_CH;
    hoff += 16 * RES_CH;
  }
}

// ---------------------------------------------------------------- post kernels

// y1 = relu(last1_w @ relu(skips) + b1), stored f16 time-major [B][T][64]
__global__ void k_last1(const float* __restrict__ skips,
                        const float* __restrict__ w1, const float* __restrict__ b1,
                        _Float16* __restrict__ y1, int B, int T) {
  const size_t total = (size_t)B * T * SKIP_CH;
  for (size_t i = (size_t)blockIdx.x * blockDim.x + threadIdx.x; i < total;
       i += (size_t)gridDim.x * blockDim.x) {
    int o = (int)(i & 63);
    size_t bt = i >> 6;
    const float* sp = skips + bt * SKIP_CH;
    float acc = b1[o];
    for (int k = 0; k < SKIP_CH; ++k)
      acc += w1[o * SKIP_CH + k] * fmaxf(sp[k], 0.0f);
    y1[i] = (_Float16)fmaxf(acc, 0.0f);
  }
}

// out[b][o][t] = last2_w @ y1 + b2, WMMA over 16 out-channel x 16 time tiles
__global__ __launch_bounds__(256) void k_last2(
    const _Float16* __restrict__ y1, const float* __restrict__ w2,
    const float* __restrict__ b2, float* __restrict__ out, int B, int T) {
  const int tid = threadIdx.x, lane = tid & 31, w = tid >> 5;
  const int b = blockIdx.y;
  const int t0 = blockIdx.x * 64;
  const int n = lane & 15, mb = (lane >> 4) * 8, kb = (lane >> 4) * 16;

  // wave handles out-channel tiles w and w+8
  const v16h a00 = load_a_f32(w2 + (size_t)(w * 16) * 64, 64, 1, 32);
  const v16h a01 = load_a_f32(w2 + (size_t)(w * 16) * 64 + 32, 64, 1, 32);
  const v16h a10 = load_a_f32(w2 + (size_t)((w + 8) * 16) * 64, 64, 1, 32);
  const v16h a11 = load_a_f32(w2 + (size_t)((w + 8) * 16) * 64 + 32, 64, 1, 32);
  float bv0[8], bv1[8];
#pragma unroll
  for (int r = 0; r < 8; ++r) {
    bv0[r] = b2[w * 16 + mb + r];
    bv1[r] = b2[(w + 8) * 16 + mb + r];
  }

  const _Float16* py = y1 + ((size_t)b * T + t0 + n) * 64 + kb;
  for (int ts = 0; ts < 4; ++ts) {
    const v16h bf0 = *(const v16h*)(py);
    const v16h bf1 = *(const v16h*)(py + 32);
    __builtin_amdgcn_sched_barrier(0);
    v8f d0 = vzero8(), d1 = vzero8();
    d0 = wmma16(a00, bf0, d0);
    d0 = wmma16(a01, bf1, d0);
    d1 = wmma16(a10, bf0, d1);
    d1 = wmma16(a11, bf1, d1);
    const int tg = t0 + ts * 16 + n;
#pragma unroll
    for (int r = 0; r < 8; ++r) {
      out[((size_t)b * OUT_CH + w * 16 + mb + r) * T + tg] = d0[r] + bv0[r];
      out[((size_t)b * OUT_CH + (w + 8) * 16 + mb + r) * T + tg] = d1[r] + bv1[r];
    }
    py += 16 * 64;
  }
}

// ---------------------------------------------------------------- launch

extern "C" void kernel_launch(void* const* d_in, const int* in_sizes, int n_in,
                              void* d_out, int out_size, void* d_ws, size_t ws_size,
                              hipStream_t stream) {
  const int*   x         = (const int*)d_in[0];
  const float* condition = (const float*)d_in[1];
  const float* first_w   = (const float*)d_in[2];
  const float* first_b   = (const float*)d_in[3];
  const float* cin_w     = (const float*)d_in[4];
  const float* conv_w    = (const float*)d_in[5];
  const float* conv_b    = (const float*)d_in[6];
  const float* cond_w    = (const float*)d_in[7];
  const float* skip_w    = (const float*)d_in[8];
  const float* skip_b    = (const float*)d_in[9];
  const float* out_w     = (const float*)d_in[10];
  const float* out_b     = (const float*)d_in[11];
  const float* last1_w   = (const float*)d_in[12];
  const float* last1_b   = (const float*)d_in[13];
  const float* last2_w   = (const float*)d_in[14];
  const float* last2_b   = (const float*)d_in[15];
  float* out = (float*)d_out;
  (void)in_sizes; (void)n_in; (void)out_size; (void)ws_size;

  const int B = 8, Tf = 400, T = 32000;

  char* p = (char*)d_ws;
  auto take = [&](size_t bytes) -> char* {
    char* r = p;
    p += (bytes + 255) & ~(size_t)255;
    return r;
  };
  _Float16* h0    = (_Float16*)take((size_t)B * (TPAD + T) * RES_CH * sizeof(_Float16));
  _Float16* h1    = (_Float16*)take((size_t)B * (TPAD + T) * RES_CH * sizeof(_Float16));
  _Float16* cf16  = (_Float16*)take((size_t)B * T * CIN_PAD * sizeof(_Float16));
  float*    skips = (float*)take((size_t)B * T * SKIP_CH * sizeof(float));
  _Float16* y1    = (_Float16*)take((size_t)B * T * SKIP_CH * sizeof(_Float16));
  float*    c1    = (float*)take((size_t)B * CIN_CH * Tf * sizeof(float));
  float*    c2    = (float*)take((size_t)B * CIN_CH * Tf * 10 * sizeof(float));
  _Float16* wpack = (_Float16*)take((size_t)LAYERS * 8 * NTILE * 512 * sizeof(_Float16));

  auto blocks = [](size_t total) {
    return dim3((unsigned)((total + 255) / 256));
  };

  k_pack<<<blocks((size_t)LAYERS * 8 * NTILE * 32), 256, 0, stream>>>(
      conv_w, cond_w, skip_w, out_w, wpack);
  k_zero_hpad<<<blocks((size_t)B * TPAD * RES_CH), 256, 0, stream>>>(h0, B, T);
  k_zero_hpad<<<blocks((size_t)B * TPAD * RES_CH), 256, 0, stream>>>(h1, B, T);
  k_zero_f32<<<blocks((size_t)B * T * SKIP_CH), 256, 0, stream>>>(
      skips, (size_t)B * T * SKIP_CH);
  k_zero_f16<<<blocks((size_t)B * T * CIN_PAD), 256, 0, stream>>>(
      cf16, (size_t)B * T * CIN_PAD);
  k_embed<<<blocks((size_t)B * T), 256, 0, stream>>>(x, first_w, first_b, h0, B, T);
  k_cond_proj<<<blocks((size_t)B * CIN_CH * Tf), 256, 0, stream>>>(
      condition, cin_w, c1, B, Tf);
  k_upsample_f32<<<blocks((size_t)B * CIN_CH * Tf * 10), 256, 0, stream>>>(
      c1, c2, B, CIN_CH, Tf, 10);
  k_upsample_f16pad<<<blocks((size_t)B * CIN_CH * T), 256, 0, stream>>>(
      c2, cf16, B, CIN_CH, Tf * 10, 8);

  dim3 lgrid(T / TT, B);
  const _Float16* hin = h0;
  _Float16* hout = h1;
  for (int l = 0; l < LAYERS; ++l) {
    k_layer<<<lgrid, 256, 0, stream>>>(hin, hout, cf16, skips, wpack, conv_b,
                                       skip_b, out_b, l, T);
    const _Float16* t2 = hout;
    hout = (_Float16*)hin;
    hin = t2;
  }

  k_last1<<<blocks((size_t)B * T * SKIP_CH), 256, 0, stream>>>(
      skips, last1_w, last1_b, y1, B, T);
  dim3 l2grid(T / 64, B);
  k_last2<<<l2grid, 256, 0, stream>>>(y1, last2_w, last2_b, out, B, T);
}
